// PGATActor_46084999086427
// MI455X (gfx1250) — matrix operand order repaired and checked by reading, not compiled, not run on hardware
//
#include <hip/hip_runtime.h>
#include <math.h>

// ---------------- problem constants (from reference) ----------------
#define NN     16384          // B*A nodes
#define HEADS  4
#define CDIM   128            // C = HID
#define FDIM   512            // HEADS*C
#define HID_   128
#define NOUT_  2
#define SCALE_ 0.17677669529663687f   // (HID/HEADS)^-0.5 = 1/sqrt(32)

typedef unsigned short ushort_t;
typedef __attribute__((ext_vector_type(16))) __bf16 v16bf;
typedef __attribute__((ext_vector_type(8)))  float  v8f;
typedef __attribute__((ext_vector_type(4)))  unsigned int u32x4;
typedef __attribute__((ext_vector_type(4)))  float  f32x4;

static inline int cdiv(int a, int b) { return (a + b - 1) / b; }

// ---------------- CDNA5 async global->LDS staging (ASYNCcnt path) ----------------
#if __has_builtin(__builtin_amdgcn_global_load_async_to_lds_b128)
#define HAVE_ASYNC_LDS 1
#else
#define HAVE_ASYNC_LDS 0
#endif

// builtin signature (from hipcc diagnostic): param0 = v4i* in AS(1) (prints as __device__)
typedef int i32v4_t __attribute__((vector_size(16)));
typedef __attribute__((address_space(1))) i32v4_t g_i32v4;   // global
typedef __attribute__((address_space(3))) i32v4_t l_i32v4;   // LDS

// copy 16 bytes global -> LDS; async engine if available, VGPR round-trip otherwise
__device__ __forceinline__ void stage_b128(const ushort_t* g, ushort_t* l) {
#if HAVE_ASYNC_LDS
    __builtin_amdgcn_global_load_async_to_lds_b128(
        (g_i32v4*)(unsigned long long)g,
        (l_i32v4*)(unsigned int)(unsigned long long)l,   // flat addr[31:0] == LDS offset
        0, 0);
#else
    *(u32x4*)l = *(const u32x4*)g;
#endif
}

__device__ __forceinline__ void stage_wait() {
#if HAVE_ASYNC_LDS
#if __has_builtin(__builtin_amdgcn_s_wait_asynccnt)
    __builtin_amdgcn_s_wait_asynccnt(0);
#else
    asm volatile("s_wait_asynccnt 0" ::: "memory");
#endif
#endif
}

// ---------------- utility kernels ----------------
__global__ void fill_f32_kernel(float* __restrict__ p, float v, int n) {
    int i = blockIdx.x * 256 + threadIdx.x;
    if (i < n) p[i] = v;
}

// fp32 -> bf16 (round to nearest even)
__global__ void cvt_bf16_kernel(const float* __restrict__ in, ushort_t* __restrict__ out, int n) {
    int i = blockIdx.x * 256 + threadIdx.x;
    if (i < n) {
        unsigned u = __float_as_uint(in[i]);
        u += 0x7FFFu + ((u >> 16) & 1u);
        out[i] = (ushort_t)(u >> 16);
    }
}

// ---------------- attention: Gram matrix of Wq/Wk over channel dim ----------------
// G[h][a][b] = sum_c Wq[h*C+c, a] * Wk[h*C+c, b]   (a,b in {0,1})
__global__ void gram_kernel(const float* __restrict__ Wq, const float* __restrict__ Wk,
                            float* __restrict__ G) {
    int t = threadIdx.x;
    if (t >= HEADS * 4) return;
    int h = t >> 2, a = (t >> 1) & 1, b = t & 1;
    float s = 0.f;
    for (int c = 0; c < CDIM; ++c) {
        int j = h * CDIM + c;
        s += Wq[2 * j + a] * Wk[2 * j + b];
    }
    G[t] = s;
}

__device__ __forceinline__ void atomicMaxF(float* addr, float val) {
    if (val >= 0.f) atomicMax((int*)addr, __float_as_int(val));
    else            atomicMin((unsigned int*)addr, __float_as_uint(val));
}

// per (edge, head): alpha = SCALE * pd^T G ps ; segment max into amax[dst]
__global__ void edge_alpha_kernel(const float* __restrict__ pos,
                                  const int* __restrict__ src, const int* __restrict__ dst,
                                  const float* __restrict__ G,
                                  float* __restrict__ alpha, float* __restrict__ amax, int E) {
    int t = blockIdx.x * 256 + threadIdx.x;
    if (t >= E * HEADS) return;
    int e = t >> 2, h = t & 3;
    int s = src[e], d = dst[e];
    float ps0 = pos[2 * s], ps1 = pos[2 * s + 1];
    float pd0 = pos[2 * d], pd1 = pos[2 * d + 1];
    const float* g = G + 4 * h;
    float a = (pd0 * ps0 * g[0] + pd0 * ps1 * g[1] +
               pd1 * ps0 * g[2] + pd1 * ps1 * g[3]) * SCALE_;
    alpha[t] = a;
    atomicMaxF(amax + d * HEADS + h, a);
}

// ex = exp(alpha - amax[dst]) in place ; segment sum into asum[dst]
__global__ void edge_exp_kernel(float* __restrict__ alpha, const float* __restrict__ amax,
                                float* __restrict__ asum, const int* __restrict__ dst, int E) {
    int t = blockIdx.x * 256 + threadIdx.x;
    if (t >= E * HEADS) return;
    int e = t >> 2, h = t & 3;
    int d = dst[e];
    float ex = __expf(alpha[t] - amax[d * HEADS + h]);
    alpha[t] = ex;
    atomicAdd(asum + d * HEADS + h, ex);
}

// agg[dst] += (ex/asum[dst]) * v[src] ; one thread handles 4 channels
__global__ void edge_agg_kernel(const float* __restrict__ v, const float* __restrict__ ex,
                                const float* __restrict__ asum,
                                const int* __restrict__ src, const int* __restrict__ dst,
                                float* __restrict__ agg, int E) {
    int t = blockIdx.x * 256 + threadIdx.x;
    if (t >= E * (FDIM / 4)) return;
    int e  = t >> 7;          // FDIM/4 = 128 chunks per edge
    int j0 = (t & 127) * 4;   // channel base in [0, 512)
    int h  = j0 >> 7;         // head = channel/128
    int s = src[e], d = dst[e];
    float a = ex[e * HEADS + h] / asum[d * HEADS + h];
    f32x4 vv = *(const f32x4*)(v + (size_t)s * FDIM + j0);
    float* ap = agg + (size_t)d * FDIM + j0;
    atomicAdd(ap + 0, vv.x * a);
    atomicAdd(ap + 1, vv.y * a);
    atomicAdd(ap + 2, vv.z * a);
    atomicAdd(ap + 3, vv.w * a);
}

// ---------------- bf16 WMMA GEMM: out = act(A(MxK) @ W(NxK)^T + bias) ----------------
// Block = 256 threads (8 waves) -> 128(M) x 64(N) output tile.
//   * B tile (64 x 32k bf16) double-buffered in LDS via async global->LDS copies;
//     the copy of chunk kb+32 overlaps the 4 WMMAs of chunk kb (one barrier per chunk).
//   * Each wave owns a 16(M) x 64(N) strip: one A-frag feeds 4 WMMAs (4 accumulators).
// Requires M % 128 == 0, N % 64 == 0, K % 32 == 0 (true for every GEMM here).
// Fragment layouts (ISA 7.12.2, wave32):
//   A (16x32 bf16): lane = (m%16) + 16*hi ; halfs 0-7 -> K=hi*8+0..7, 8-15 -> K=16+hi*8+0..7
//   B (32x16 bf16): lane = (n%16) + 16*hi ; halfs 0-7 -> K=hi*8+0..7, 8-15 -> K=16+hi*8+0..7
//   C/D (16x16 f32): vgpr r, lanes 0-15 -> (M=r, N=lane), lanes 16-31 -> (M=8+r, N=lane-16)
template <int ACT>   // 0=none, 1=tanh, 2=relu
__global__ void __launch_bounds__(256)
gemm_bf16_kernel(const ushort_t* __restrict__ A, const ushort_t* __restrict__ W,
                 const float* __restrict__ bias, float* __restrict__ out,
                 int M, int N, int K) {
    __shared__ __align__(16) ushort_t Bs[2][64 * 32];   // 2 x 4 KB double buffer

    const int tid  = threadIdx.x;
    const int lane = tid & 31;
    const int wave = tid >> 5;
    const int nb64 = N >> 6;
    const int bm   = blockIdx.x / nb64;
    const int bn   = blockIdx.x - bm * nb64;
    const int m0   = bm * 128 + wave * 16;
    const int n0   = bn * 64;

    const int half = lane >> 4;   // hi half of the wave
    const int l15  = lane & 15;

    const ushort_t* Arow = A + (size_t)(m0 + l15) * K + half * 8;

    // cooperative B staging coords: thread -> (row 0..63, 8 bf16 chunk)
    const int sr = tid >> 2;           // 0..63
    const int sc = (tid & 3) * 8;      // 0,8,16,24
    const int soff = sr * 32 + sc;
    const ushort_t* Wsrc = W + (size_t)(n0 + sr) * K + sc;

    v8f acc[4];
#pragma unroll
    for (int nf = 0; nf < 4; ++nf)
        acc[nf] = (v8f){0.f, 0.f, 0.f, 0.f, 0.f, 0.f, 0.f, 0.f};

    // prologue: stage K-chunk 0 into buffer 0
    stage_b128(Wsrc, &Bs[0][soff]);
    stage_wait();
    __syncthreads();

    int buf = 0;
    for (int kb = 0; kb < K; kb += 32, buf ^= 1) {
        // kick off next chunk's copy into the other buffer (overlaps WMMAs below);
        // safe: everyone finished reading buf^1 at the barrier one iteration ago
        if (kb + 32 < K) {
            stage_b128(Wsrc + kb + 32, &Bs[buf ^ 1][soff]);
            __builtin_prefetch((const void*)(Arow + kb + 32), 0, 3);
        }

        union { u32x4 u[2]; v16bf v; } af;
        af.u[0] = *(const u32x4*)(Arow + kb);              // K = kb + hi*8 + 0..7
        af.u[1] = *(const u32x4*)(Arow + kb + 16);         // K = kb + 16 + hi*8 + 0..7

        const ushort_t* Bbase = &Bs[buf][0];
#pragma unroll
        for (int nf = 0; nf < 4; ++nf) {
            union { u32x4 u[2]; v16bf v; } bf;
            const ushort_t* bsrc = Bbase + (nf * 16 + l15) * 32 + half * 8;
            bf.u[0] = *(const u32x4*)(bsrc);               // K = kb + hi*8 + 0..7
            bf.u[1] = *(const u32x4*)(bsrc + 16);          // K = kb + 16 + hi*8 + 0..7
            acc[nf] = __builtin_amdgcn_wmma_f32_16x16x32_bf16(
                /*neg_a=*/false, af.v, /*neg_b=*/false, bf.v,
                /*c_mod=*/(short)0, acc[nf], /*reuse_a=*/false, /*reuse_b=*/false);
        }

        stage_wait();        // next buffer's DMA landed (this wave's share)
        __syncthreads();     // visible to all waves; all reads of Bs[buf] complete
    }

#pragma unroll
    for (int nf = 0; nf < 4; ++nf) {
        const int nc = n0 + nf * 16 + l15;
        const float bval = bias ? bias[nc] : 0.f;
#pragma unroll
        for (int r = 0; r < 8; ++r) {
            float val = acc[nf][r] + bval;
            if (ACT == 1) val = tanhf(val);
            if (ACT == 2) val = fmaxf(val, 0.f);
            out[(size_t)(m0 + half * 8 + r) * N + nc] = val;
        }
    }
}

// ---------------- final 256 -> 2 projection (too narrow for WMMA) ----------------
__global__ void final_kernel(const float* __restrict__ h, const float* __restrict__ W3,
                             const float* __restrict__ b3, float* __restrict__ out, int M) {
    int t = blockIdx.x * 256 + threadIdx.x;
    if (t >= M * NOUT_) return;
    int n = t >> 1, o = t & 1;
    const float* hr = h + (size_t)n * 256;
    const float* wr = W3 + o * 256;
    float s = b3[o];
    for (int k = 0; k < 256; ++k) s += hr[k] * wr[k];
    out[t] = s;
}

// ---------------- host orchestration ----------------
extern "C" void kernel_launch(void* const* d_in, const int* in_sizes, int n_in,
                              void* d_out, int out_size, void* d_ws, size_t ws_size,
                              hipStream_t stream) {
    const float* x     = (const float*)d_in[0];
    const float* pos   = (const float*)d_in[1];
    const int*   eidx  = (const int*)d_in[2];
    const float* Wq0   = (const float*)d_in[3];
    const float* Wk0   = (const float*)d_in[4];
    const float* Wv0   = (const float*)d_in[5];
    const float* Wo0   = (const float*)d_in[6];
    const float* bo0   = (const float*)d_in[7];
    const float* Wq1   = (const float*)d_in[8];
    const float* Wk1   = (const float*)d_in[9];
    const float* Wv1   = (const float*)d_in[10];
    const float* Wo1   = (const float*)d_in[11];
    const float* bo1   = (const float*)d_in[12];
    const float* W1    = (const float*)d_in[13];
    const float* b1    = (const float*)d_in[14];
    const float* W2    = (const float*)d_in[15];
    const float* b2    = (const float*)d_in[16];
    const float* W3    = (const float*)d_in[17];
    const float* b3    = (const float*)d_in[18];
    float* out = (float*)d_out;

    const int N = NN;
    const int E = in_sizes[2] / 2;
    const int* src = eidx;
    const int* dst = eidx + E;

    // workspace carve-up
    char* w = (char*)d_ws;
    float*    G     = (float*)w;                 w += 32 * sizeof(float);      // G0 | G1
    float*    alpha = (float*)w;                 w += (size_t)E * HEADS * 4;   // alpha / ex
    float*    amax  = (float*)w;                 w += (size_t)N * HEADS * 4;
    float*    asum  = (float*)w;                 w += (size_t)N * HEADS * 4;
    float*    vbuf  = (float*)w;                 w += (size_t)N * FDIM * 4;    // also MLP h2
    float*    agg   = (float*)w;                 w += (size_t)N * FDIM * 4;    // also MLP h3
    float*    hbuf  = (float*)w;                 w += (size_t)N * HID_ * 4;
    ushort_t* abf   = (ushort_t*)w;              w += (size_t)N * FDIM * 2;    // bf16 activations
    ushort_t* wbf   = (ushort_t*)w;              /* 128KB max weights */       // bf16 weights

    const float NEG_INF = -INFINITY;

    auto launch_gemm = [&](int act, const ushort_t* Ab, const ushort_t* Wb,
                           const float* bias, float* O, int M, int Nc, int K) {
        int blocks = (M / 128) * (Nc / 64);
        if (act == 0) gemm_bf16_kernel<0><<<blocks, 256, 0, stream>>>(Ab, Wb, bias, O, M, Nc, K);
        if (act == 1) gemm_bf16_kernel<1><<<blocks, 256, 0, stream>>>(Ab, Wb, bias, O, M, Nc, K);
        if (act == 2) gemm_bf16_kernel<2><<<blocks, 256, 0, stream>>>(Ab, Wb, bias, O, M, Nc, K);
    };

    // ======== GAT layer 0 ========
    gram_kernel<<<1, 32, 0, stream>>>(Wq0, Wk0, G);
    fill_f32_kernel<<<cdiv(N * HEADS, 256), 256, 0, stream>>>(amax, NEG_INF, N * HEADS);
    fill_f32_kernel<<<cdiv(N * HEADS, 256), 256, 0, stream>>>(asum, 0.f, N * HEADS);
    fill_f32_kernel<<<cdiv(N * FDIM, 256), 256, 0, stream>>>(agg, 0.f, N * FDIM);

    cvt_bf16_kernel<<<cdiv(N * 64, 256), 256, 0, stream>>>(x, abf, N * 64);
    cvt_bf16_kernel<<<cdiv(FDIM * 64, 256), 256, 0, stream>>>(Wv0, wbf, FDIM * 64);
    launch_gemm(0, abf, wbf, nullptr, vbuf, N, FDIM, 64);               // v0 = x @ Wv0^T

    edge_alpha_kernel<<<cdiv(E * HEADS, 256), 256, 0, stream>>>(pos, src, dst, G, alpha, amax, E);
    edge_exp_kernel<<<cdiv(E * HEADS, 256), 256, 0, stream>>>(alpha, amax, asum, dst, E);
    edge_agg_kernel<<<cdiv(E * (FDIM / 4), 256), 256, 0, stream>>>(vbuf, alpha, asum, src, dst, agg, E);

    cvt_bf16_kernel<<<cdiv(N * FDIM, 256), 256, 0, stream>>>(agg, abf, N * FDIM);
    cvt_bf16_kernel<<<cdiv(HID_ * FDIM, 256), 256, 0, stream>>>(Wo0, wbf, HID_ * FDIM);
    launch_gemm(1, abf, wbf, bo0, hbuf, N, HID_, FDIM);                 // h = tanh(agg @ Wo0^T + bo0)

    // ======== GAT layer 1 ========
    gram_kernel<<<1, 32, 0, stream>>>(Wq1, Wk1, G + 16);
    fill_f32_kernel<<<cdiv(N * HEADS, 256), 256, 0, stream>>>(amax, NEG_INF, N * HEADS);
    fill_f32_kernel<<<cdiv(N * HEADS, 256), 256, 0, stream>>>(asum, 0.f, N * HEADS);
    fill_f32_kernel<<<cdiv(N * FDIM, 256), 256, 0, stream>>>(agg, 0.f, N * FDIM);

    cvt_bf16_kernel<<<cdiv(N * HID_, 256), 256, 0, stream>>>(hbuf, abf, N * HID_);
    cvt_bf16_kernel<<<cdiv(FDIM * HID_, 256), 256, 0, stream>>>(Wv1, wbf, FDIM * HID_);
    launch_gemm(0, abf, wbf, nullptr, vbuf, N, FDIM, HID_);             // v1 = h @ Wv1^T

    edge_alpha_kernel<<<cdiv(E * HEADS, 256), 256, 0, stream>>>(pos, src, dst, G + 16, alpha, amax, E);
    edge_exp_kernel<<<cdiv(E * HEADS, 256), 256, 0, stream>>>(alpha, amax, asum, dst, E);
    edge_agg_kernel<<<cdiv(E * (FDIM / 4), 256), 256, 0, stream>>>(vbuf, alpha, asum, src, dst, agg, E);

    cvt_bf16_kernel<<<cdiv(N * FDIM, 256), 256, 0, stream>>>(agg, abf, N * FDIM);
    cvt_bf16_kernel<<<cdiv(HID_ * FDIM, 256), 256, 0, stream>>>(Wo1, wbf, HID_ * FDIM);
    launch_gemm(1, abf, wbf, bo1, hbuf, N, HID_, FDIM);                 // h = tanh(agg @ Wo1^T + bo1)

    // ======== MLP head ========
    cvt_bf16_kernel<<<cdiv(N * HID_, 256), 256, 0, stream>>>(hbuf, abf, N * HID_);
    cvt_bf16_kernel<<<cdiv(256 * HID_, 256), 256, 0, stream>>>(W1, wbf, 256 * HID_);
    launch_gemm(2, abf, wbf, b1, vbuf, N, 256, HID_);                   // h2 = relu(h @ W1^T + b1)

    cvt_bf16_kernel<<<cdiv(N * 256, 256), 256, 0, stream>>>(vbuf, abf, N * 256);
    cvt_bf16_kernel<<<cdiv(256 * 256, 256), 256, 0, stream>>>(W2, wbf, 256 * 256);
    launch_gemm(2, abf, wbf, b2, agg, N, 256, 256);                     // h3 = relu(h2 @ W2^T + b2)

    final_kernel<<<cdiv(N * NOUT_, 256), 256, 0, stream>>>(agg, W3, b3, out, N);
}
